// EGNN_Layer_55946243998163
// MI455X (gfx1250) — matrix operand order repaired and checked by reading, compiled, not run on hardware
//
#include <hip/hip_runtime.h>

#define HD 128

typedef __attribute__((ext_vector_type(16))) _Float16 v16h;
typedef __attribute__((ext_vector_type(8)))  _Float16 v8h;
typedef __attribute__((ext_vector_type(8)))  float    v8f;

__device__ __forceinline__ float silu_f(float v) {
    return v / (1.0f + __expf(-v));
}

// A-fragment (16-bit A 16x32 layout, contiguous K to match B fragments):
//  lanes 0-15 : M=lane,    K = kt*32 + {0..15}
//  lanes 16-31: M=lane-16, K = kt*32 + {16..31}
// => per lane: two contiguous 8-half (16B) chunks from LDS.
__device__ __forceinline__ v16h lds_a_frag(const _Float16* sbuf, int m, int kdim,
                                           int kt, int r8) {
    const v8h lo = *(const v8h*)(sbuf + m * kdim + kt * 32 + r8 * 2);
    const v8h hi = *(const v8h*)(sbuf + m * kdim + kt * 32 + r8 * 2 + 8);
    return __builtin_shufflevector(lo, hi, 0, 1, 2, 3, 4, 5, 6, 7,
                                           8, 9, 10, 11, 12, 13, 14, 15);
}

// Pre-swizzled B fragment: lane holds N = nt*16 + (lane&15),
// K = kt*32 + (lane>>4)*16 + 0..15 -> 16 contiguous halves (32B aligned).
__device__ __forceinline__ v16h b_frag(const _Float16* frags, int kt, int nt, int lane) {
    return ((const v16h*)frags)[(kt * 8 + nt) * 32 + lane];
}

// One K-tile of a [16 x K] @ [K x 128] GEMM: batch all 8 B-fragment loads
// into registers first (one VMEM clause), then drain 8 WMMAs so the
// s_wait_loadcnt values stagger and fragment fetch overlaps matrix issue.
__device__ __forceinline__ void gemm_ktile(const _Float16* sbuf, int kdim,
                                           const _Float16* frags, int kt,
                                           int m, int r8, int lane, v8f c[8]) {
    v16h a = lds_a_frag(sbuf, m, kdim, kt, r8);
    v16h b[8];
#pragma unroll
    for (int nt = 0; nt < 8; ++nt) b[nt] = b_frag(frags, kt, nt, lane);
#pragma unroll
    for (int nt = 0; nt < 8; ++nt) {
        c[nt] = __builtin_amdgcn_wmma_f32_16x16x32_f16(false, a, false, b[nt],
                                                       (short)0, c[nt], false, false);
    }
}

// Build B fragments from a row-major [K x 128] f32 weight matrix.
__global__ void prep_frag_kernel(const float* __restrict__ W,
                                 _Float16* __restrict__ dst, int total) {
    int idx = blockIdx.x * blockDim.x + threadIdx.x;
    if (idx >= total) return;
    int i    = idx & 15;
    int lane = (idx >> 4) & 31;
    int nt   = (idx >> 9) & 7;
    int kt   = idx >> 12;
    int k = kt * 32 + ((lane >> 4) << 4) + i;
    int n = nt * 16 + (lane & 15);
    dst[idx] = (_Float16)W[k * HD + n];
}

__global__ void init_kernel(float* __restrict__ m_acc, float* __restrict__ x_out,
                            const float* __restrict__ x, int nAcc, int nX) {
    int stride = gridDim.x * blockDim.x;
    int tid = blockIdx.x * blockDim.x + threadIdx.x;
    for (int i = tid; i < nAcc; i += stride) m_acc[i] = 0.0f;
    for (int i = tid; i < nX; i += stride) x_out[i] = x[i];
}

__global__ __launch_bounds__(32) void egnn_edge_kernel(
    const float* __restrict__ h, const float* __restrict__ x,
    const int* __restrict__ ei,
    const float* __restrict__ ew1, const float* __restrict__ eb1,
    const float* __restrict__ eb2, const float* __restrict__ cb1,
    const float* __restrict__ cw2, const float* __restrict__ cb2,
    const _Float16* __restrict__ ew1f, const _Float16* __restrict__ ew2f,
    const _Float16* __restrict__ cw1f,
    float* __restrict__ m_acc, float* __restrict__ x_out, int N, int E) {
    __shared__ __align__(16) _Float16 sA[16 * 256];    // [h[row] | h[col]] f16
    __shared__ __align__(16) _Float16 sAct[16 * 128];  // edge-MLP layer1 act
    __shared__ __align__(16) _Float16 sM[16 * 128];    // m_ij (f16 copy)
    __shared__ __align__(16) float    sC[16 * 128];    // coord-MLP layer1 act
    __shared__ __align__(16) float    sCw2[128];       // cw2 broadcast copy
    __shared__ float sRad[16];
    __shared__ float sDiff[16 * 3];
    __shared__ int   sRow[16];
    __shared__ float sVal[16];

    const int lane = threadIdx.x;
    const int e0 = blockIdx.x * 16;

    // stage cw2 (128 f32) into LDS: 4 values per lane
    {
        float4 w = *(const float4*)(cw2 + lane * 4);
        *(float4*)(sCw2 + lane * 4) = w;
    }
    if (lane < 16) {
        int e = e0 + lane;
        int valid = (e < E);
        int ec = valid ? e : (E - 1);
        int r = ei[ec];
        int c = ei[E + ec];
        sRow[lane] = r;
        sVal[lane] = valid ? 1.0f : 0.0f;
        float dx = x[r * 3 + 0] - x[c * 3 + 0];
        float dy = x[r * 3 + 1] - x[c * 3 + 1];
        float dz = x[r * 3 + 2] - x[c * 3 + 2];
        sDiff[lane * 3 + 0] = dx;
        sDiff[lane * 3 + 1] = dy;
        sDiff[lane * 3 + 2] = dz;
        sRad[lane] = dx * dx + dy * dy + dz * dz;
    }
    {   // gather h[row], h[col] into the 16x256 f16 A-tile (2 lanes per edge)
        int eidx = lane >> 1;
        int part = lane & 1;
        int e = e0 + eidx;
        if (e >= E) e = E - 1;
        int src = part ? ei[E + e] : ei[e];
        const float* hp = h + (size_t)src * HD;
        _Float16* dstp = sA + eidx * 256 + part * 128;
#pragma unroll
        for (int j = 0; j < 128; j += 8) {
            float4 a0 = *(const float4*)(hp + j);
            float4 a1 = *(const float4*)(hp + j + 4);
            v8h hv;
            hv[0] = (_Float16)a0.x; hv[1] = (_Float16)a0.y;
            hv[2] = (_Float16)a0.z; hv[3] = (_Float16)a0.w;
            hv[4] = (_Float16)a1.x; hv[5] = (_Float16)a1.y;
            hv[6] = (_Float16)a1.z; hv[7] = (_Float16)a1.w;
            *(v8h*)(dstp + j) = hv;
        }
    }
    __syncthreads();

    const int m  = lane & 15;
    const int r8 = (lane >> 4) << 3;
    const v8f vzero = {0.f, 0.f, 0.f, 0.f, 0.f, 0.f, 0.f, 0.f};
    v8f c[8];

    // ---- edge MLP layer 1: [16x256]@[256x128] + radial rank-1 + bias, SiLU ----
#pragma unroll
    for (int nt = 0; nt < 8; ++nt) c[nt] = vzero;
#pragma unroll
    for (int kt = 0; kt < 8; ++kt)
        gemm_ktile(sA, 256, ew1f, kt, m, r8, lane, c);
#pragma unroll
    for (int nt = 0; nt < 8; ++nt) {
        int n = nt * 16 + m;
        float wr = ew1[256 * HD + n];  // row 257 of ew1 multiplies radial
        float bb = eb1[n];
#pragma unroll
        for (int i = 0; i < 8; ++i) {
            int M = r8 + i;
            float v = c[nt][i] + sRad[M] * wr + bb;
            sAct[M * 128 + n] = (_Float16)silu_f(v);
        }
    }
    __syncthreads();

    // ---- edge MLP layer 2: [16x128]@[128x128] + bias, SiLU -> m_ij ----
#pragma unroll
    for (int nt = 0; nt < 8; ++nt) c[nt] = vzero;
#pragma unroll
    for (int kt = 0; kt < 4; ++kt)
        gemm_ktile(sAct, 128, ew2f, kt, m, r8, lane, c);
#pragma unroll
    for (int nt = 0; nt < 8; ++nt) {
        int n = nt * 16 + m;
        float bb = eb2[n];
#pragma unroll
        for (int i = 0; i < 8; ++i) {
            int M = r8 + i;
            float v = silu_f(c[nt][i] + bb);
            sM[M * 128 + n] = (_Float16)v;
            // segment_sum(m_ij, row) via f32 atomics into workspace accumulator
            atomicAdd(&m_acc[(size_t)sRow[M] * HD + n], v * sVal[M]);
        }
    }
    __syncthreads();

    // ---- coord MLP layer 1: [16x128]@[128x128] + bias, SiLU ----
#pragma unroll
    for (int nt = 0; nt < 8; ++nt) c[nt] = vzero;
#pragma unroll
    for (int kt = 0; kt < 4; ++kt)
        gemm_ktile(sM, 128, cw1f, kt, m, r8, lane, c);
#pragma unroll
    for (int nt = 0; nt < 8; ++nt) {
        int n = nt * 16 + m;
        float bb = cb1[n];
#pragma unroll
        for (int i = 0; i < 8; ++i) {
            int M = r8 + i;
            sC[M * 128 + n] = silu_f(c[nt][i] + bb);
        }
    }
    __syncthreads();

    // ---- coord MLP layer 2 (128 -> 1) + force scatter ----
    if (lane < 16) {
        int e = e0 + lane;
        if (e < E) {
            float acc = cb2[0];
            const float* cp = sC + lane * 128;
#pragma unroll 4
            for (int k = 0; k < 128; ++k) acc += cp[k] * sCw2[k];
            int r = sRow[lane];
            atomicAdd(&x_out[r * 3 + 0], sDiff[lane * 3 + 0] * acc);
            atomicAdd(&x_out[r * 3 + 1], sDiff[lane * 3 + 1] * acc);
            atomicAdd(&x_out[r * 3 + 2], sDiff[lane * 3 + 2] * acc);
        }
    }
}

__global__ __launch_bounds__(32) void egnn_node_kernel(
    const float* __restrict__ h, const float* __restrict__ m_acc,
    const float* __restrict__ nb1, const float* __restrict__ nb2,
    const _Float16* __restrict__ nw1f, const _Float16* __restrict__ nw2f,
    float* __restrict__ h_out, int N) {
    __shared__ __align__(16) _Float16 sA[16 * 256];
    __shared__ __align__(16) _Float16 sAct[16 * 128];

    const int lane = threadIdx.x;
    const int n0 = blockIdx.x * 16;
    {
        int nidx = lane >> 1;
        int part = lane & 1;
        int node = n0 + nidx;
        if (node >= N) node = N - 1;
        const float* sp = (part ? m_acc : h) + (size_t)node * HD;
        _Float16* dstp = sA + nidx * 256 + part * 128;
#pragma unroll
        for (int j = 0; j < 128; j += 8) {
            float4 a0 = *(const float4*)(sp + j);
            float4 a1 = *(const float4*)(sp + j + 4);
            v8h hv;
            hv[0] = (_Float16)a0.x; hv[1] = (_Float16)a0.y;
            hv[2] = (_Float16)a0.z; hv[3] = (_Float16)a0.w;
            hv[4] = (_Float16)a1.x; hv[5] = (_Float16)a1.y;
            hv[6] = (_Float16)a1.z; hv[7] = (_Float16)a1.w;
            *(v8h*)(dstp + j) = hv;
        }
    }
    __syncthreads();

    const int m  = lane & 15;
    const int r8 = (lane >> 4) << 3;
    const v8f vzero = {0.f, 0.f, 0.f, 0.f, 0.f, 0.f, 0.f, 0.f};
    v8f c[8];

    // node MLP layer 1: [16x256]@[256x128] + bias, SiLU
#pragma unroll
    for (int nt = 0; nt < 8; ++nt) c[nt] = vzero;
#pragma unroll
    for (int kt = 0; kt < 8; ++kt)
        gemm_ktile(sA, 256, nw1f, kt, m, r8, lane, c);
#pragma unroll
    for (int nt = 0; nt < 8; ++nt) {
        int n = nt * 16 + m;
        float bb = nb1[n];
#pragma unroll
        for (int i = 0; i < 8; ++i) {
            int M = r8 + i;
            sAct[M * 128 + n] = (_Float16)silu_f(c[nt][i] + bb);
        }
    }
    __syncthreads();

    // node MLP layer 2 + residual
#pragma unroll
    for (int nt = 0; nt < 8; ++nt) c[nt] = vzero;
#pragma unroll
    for (int kt = 0; kt < 4; ++kt)
        gemm_ktile(sAct, 128, nw2f, kt, m, r8, lane, c);
#pragma unroll
    for (int nt = 0; nt < 8; ++nt) {
        int n = nt * 16 + m;
        float bb = nb2[n];
#pragma unroll
        for (int i = 0; i < 8; ++i) {
            int M = r8 + i;
            int node = n0 + M;
            if (node < N) {
                size_t o = (size_t)node * HD + n;
                h_out[o] = h[o] + c[nt][i] + bb;
            }
        }
    }
}

extern "C" void kernel_launch(void* const* d_in, const int* in_sizes, int n_in,
                              void* d_out, int out_size, void* d_ws, size_t ws_size,
                              hipStream_t stream) {
    const float* h   = (const float*)d_in[0];
    const float* x   = (const float*)d_in[1];
    const int*   ei  = (const int*)d_in[2];
    const float* ew1 = (const float*)d_in[3];
    const float* eb1 = (const float*)d_in[4];
    const float* ew2 = (const float*)d_in[5];
    const float* eb2 = (const float*)d_in[6];
    const float* cw1 = (const float*)d_in[7];
    const float* cb1 = (const float*)d_in[8];
    const float* cw2 = (const float*)d_in[9];
    const float* cb2 = (const float*)d_in[10];
    const float* nw1 = (const float*)d_in[11];
    const float* nb1 = (const float*)d_in[12];
    const float* nw2 = (const float*)d_in[13];
    const float* nb2 = (const float*)d_in[14];

    const int N = in_sizes[0] / HD;
    const int E = in_sizes[2] / 2;

    // workspace layout
    char* wsp = (char*)d_ws;
    size_t off = 0;
    float* m_acc = (float*)(wsp + off);
    off += (size_t)N * HD * sizeof(float);
    off = (off + 255) & ~(size_t)255;
    _Float16* ew1f = (_Float16*)(wsp + off); off += (size_t)8 * 4096 * sizeof(_Float16);
    _Float16* ew2f = (_Float16*)(wsp + off); off += (size_t)4 * 4096 * sizeof(_Float16);
    _Float16* cw1f = (_Float16*)(wsp + off); off += (size_t)4 * 4096 * sizeof(_Float16);
    _Float16* nw1f = (_Float16*)(wsp + off); off += (size_t)8 * 4096 * sizeof(_Float16);
    _Float16* nw2f = (_Float16*)(wsp + off); off += (size_t)4 * 4096 * sizeof(_Float16);

    float* h_out = (float*)d_out;
    float* x_out = h_out + (size_t)N * HD;

    // weight fragment swizzle (f32 -> f16, WMMA B layout)
    prep_frag_kernel<<<(8 * 4096 + 255) / 256, 256, 0, stream>>>(ew1, ew1f, 8 * 4096);
    prep_frag_kernel<<<(4 * 4096 + 255) / 256, 256, 0, stream>>>(ew2, ew2f, 4 * 4096);
    prep_frag_kernel<<<(4 * 4096 + 255) / 256, 256, 0, stream>>>(cw1, cw1f, 4 * 4096);
    prep_frag_kernel<<<(8 * 4096 + 255) / 256, 256, 0, stream>>>(nw1, nw1f, 8 * 4096);
    prep_frag_kernel<<<(4 * 4096 + 255) / 256, 256, 0, stream>>>(nw2, nw2f, 4 * 4096);

    // zero m_acc, seed x_out = x (every launch; d_out/d_ws are poisoned)
    init_kernel<<<2048, 256, 0, stream>>>(m_acc, x_out, x, N * HD, N * 3);

    egnn_edge_kernel<<<(E + 15) / 16, 32, 0, stream>>>(
        h, x, ei, ew1, eb1, eb2, cb1, cw2, cb2, ew1f, ew2f, cw1f, m_acc, x_out, N, E);

    egnn_node_kernel<<<(N + 15) / 16, 32, 0, stream>>>(
        h, m_acc, nb1, nb2, nw1f, nw2f, h_out, N);
}
